// tDPMPN_80805514707061
// MI455X (gfx1250) — matrix operand order repaired and checked by reading, not compiled, not run on hardware
//
#include <hip/hip_runtime.h>

// Problem constants (from reference)
#define BQ   64
#define EPER 4096
#define ETOT (BQ * EPER)     // 262144 edges
#define NN   65536           // nodes
#define DD   128             // feature dim

typedef __attribute__((ext_vector_type(16))) _Float16 v16h;
typedef __attribute__((ext_vector_type(8)))  _Float16 v8h;
typedef __attribute__((ext_vector_type(8)))  float    v8f;

static __device__ __forceinline__ v16h cat16(v8h lo, v8h hi) {
  return __builtin_shufflevector(lo, hi, 0,1,2,3,4,5,6,7,8,9,10,11,12,13,14,15);
}
// Monotone float <-> uint mapping (order-preserving, invertible)
static __device__ __forceinline__ unsigned f2ord(float f) {
  unsigned u = __float_as_uint(f);
  return (u & 0x80000000u) ? ~u : (u | 0x80000000u);
}
static __device__ __forceinline__ float ord2f(unsigned u) {
  unsigned b = (u & 0x80000000u) ? (u & 0x7FFFFFFFu) : ~u;
  return __uint_as_float(b);
}
static __device__ __forceinline__ void st4h(_Float16* dst, float4 v) {
  dst[0] = (_Float16)v.x; dst[1] = (_Float16)v.y;
  dst[2] = (_Float16)v.z; dst[3] = (_Float16)v.w;
}

// ---------------------------------------------------------------------------
// Pack f32 weight matrix (Krows x ncols) into f16 WMMA B-fragments.
// Fragment frag = nt*nkt + kt holds the 32x16 tile (k0=kt*32, n0=nt*16).
// Within a fragment: lane l (N = l&15, khalf = l>>4) holds 16 contiguous f16:
//   elem j  ->  W[k0 + (j<8 ? khalf*8 + j : 16 + khalf*8 + (j-8))][n0 + (l&15)]
// so the consumer does one 32-byte contiguous load per lane.
// ---------------------------------------------------------------------------
__global__ void k_pack(const float* __restrict__ W, _Float16* __restrict__ P,
                       int ncols, int nkt, int total) {
  int idx = blockIdx.x * blockDim.x + threadIdx.x;
  if (idx >= total) return;
  int j    = idx & 15;
  int lane = (idx >> 4) & 31;
  int frag = idx >> 9;
  int kt = frag % nkt;
  int nt = frag / nkt;
  int K = kt * 32 + (j < 8 ? (lane >> 4) * 8 + j : 16 + (lane >> 4) * 8 + (j - 8));
  int N = nt * 16 + (lane & 15);
  P[idx] = (_Float16)W[(size_t)K * ncols + N];
}

// Zero output + softmax stats
__global__ void k_init(float* __restrict__ out, unsigned* __restrict__ mx,
                       float* __restrict__ den) {
  int i = blockIdx.x * blockDim.x + threadIdx.x;
  if (i < NN) { out[i] = 0.0f; mx[i] = 0u; den[i] = 0.0f; }
}

// ---------------------------------------------------------------------------
// Logits: one wave per 16-edge tile.
//   left_x/right_x (16x512 f16) staged in LDS, l/r (16x256 f16) in LDS.
//   l = lrelu(xL@Wl+bl); r = lrelu(xR@Wr+br); c = r@Wc+bc; logit = sum(l*c).
// A-fragment layout per ISA 16-bit 16x32 table; C layout: VGPR g -> row g(+8).
// ---------------------------------------------------------------------------
__global__ __launch_bounds__(32)
void k_logits(const int* __restrict__ edges,
              const float* __restrict__ node_rep,
              const float* __restrict__ rel_emb,
              const float* __restrict__ qsrc,
              const float* __restrict__ qrel,
              const _Float16* __restrict__ WlP, const float* __restrict__ bl,
              const _Float16* __restrict__ WrP, const float* __restrict__ br,
              const _Float16* __restrict__ WcP, const float* __restrict__ bc,
              float* __restrict__ logits) {
  __shared__ _Float16 xL[16 * 512];
  __shared__ _Float16 xR[16 * 512];
  __shared__ _Float16 l16[16 * 256];
  __shared__ _Float16 r16[16 * 256];

  const int lane = threadIdx.x;
  const int tile = blockIdx.x;

  // ---- stage: 2 lanes per edge, each handles 64 floats of each segment ----
  {
    int row  = lane >> 1;
    int half = lane & 1;
    int e    = tile * 16 + row;
    const int* ed = edges + (size_t)e * 8;
    int eg = ed[0], ni = ed[6], nj = ed[7];
    const float* hv = node_rep + (size_t)ni * DD + half * 64;
    const float* hw = node_rep + (size_t)nj * DD + half * 64;
    const float* re = rel_emb  + (size_t)e  * DD + half * 64;
    const float* qs = qsrc + (size_t)eg * DD + half * 64;
    const float* qr = qrel + (size_t)eg * DD + half * 64;
    int rb = row * 512 + half * 64;
#pragma unroll 4
    for (int j = 0; j < 64; j += 4) {
      st4h(&xL[rb + j], *(const float4*)(hv + j));
      st4h(&xR[rb + j], *(const float4*)(hw + j));
      float4 vr = *(const float4*)(re + j);
      st4h(&xL[rb + 128 + j], vr); st4h(&xR[rb + 128 + j], vr);
      float4 vs = *(const float4*)(qs + j);
      st4h(&xL[rb + 256 + j], vs); st4h(&xR[rb + 256 + j], vs);
      float4 vq = *(const float4*)(qr + j);
      st4h(&xL[rb + 384 + j], vq); st4h(&xR[rb + 384 + j], vq);
    }
  }
  __syncthreads();

  const int n     = lane & 15;   // B/C column within tile
  const int khalf = lane >> 4;   // A/B K-half select
  const int M     = lane & 15;   // A row
  const int mbase = khalf * 8;   // C row base for this lane group

  // ---- l and r GEMMs: 16 N-tiles x 16 K-steps, two WMMAs per step ----
  for (int nt = 0; nt < 16; ++nt) {
    v8f accL = {}; v8f accR = {};
    for (int kt = 0; kt < 16; ++kt) {
      int k0 = kt * 32;
      v8h aL0 = *(const v8h*)&xL[M * 512 + k0 + khalf * 8];
      v8h aL1 = *(const v8h*)&xL[M * 512 + k0 + 16 + khalf * 8];
      v8h aR0 = *(const v8h*)&xR[M * 512 + k0 + khalf * 8];
      v8h aR1 = *(const v8h*)&xR[M * 512 + k0 + 16 + khalf * 8];
      const v8h* pbl = (const v8h*)&WlP[(size_t)(nt * 16 + kt) * 512 + lane * 16];
      const v8h* pbr = (const v8h*)&WrP[(size_t)(nt * 16 + kt) * 512 + lane * 16];
      __builtin_prefetch(&WlP[(size_t)(nt * 16 + kt + 1) * 512 + lane * 16], 0, 1);
      v16h aLv = cat16(aL0, aL1), aRv = cat16(aR0, aR1);
      v16h bLv = cat16(pbl[0], pbl[1]), bRv = cat16(pbr[0], pbr[1]);
      accL = __builtin_amdgcn_wmma_f32_16x16x32_f16(false, aLv, false, bLv,
                                                    (short)0, accL, false, false);
      accR = __builtin_amdgcn_wmma_f32_16x16x32_f16(false, aRv, false, bRv,
                                                    (short)0, accR, false, false);
    }
    float blv = bl[nt * 16 + n];
    float brv = br[nt * 16 + n];
#pragma unroll
    for (int g = 0; g < 8; ++g) {
      float x = accL[g] + blv; x = (x > 0.0f) ? x : 0.01f * x;
      float y = accR[g] + brv; y = (y > 0.0f) ? y : 0.01f * y;
      l16[(mbase + g) * 256 + nt * 16 + n] = (_Float16)x;
      r16[(mbase + g) * 256 + nt * 16 + n] = (_Float16)y;
    }
  }
  __syncthreads();

  // ---- c = r @ Wc + bc, fused row-dot with l ----
  float acc8[8];
#pragma unroll
  for (int g = 0; g < 8; ++g) acc8[g] = 0.0f;

  for (int nt = 0; nt < 16; ++nt) {
    v8f c = {};
    for (int kt = 0; kt < 8; ++kt) {
      int k0 = kt * 32;
      v8h a0 = *(const v8h*)&r16[M * 256 + k0 + khalf * 8];
      v8h a1 = *(const v8h*)&r16[M * 256 + k0 + 16 + khalf * 8];
      const v8h* pb = (const v8h*)&WcP[(size_t)(nt * 8 + kt) * 512 + lane * 16];
      v16h av = cat16(a0, a1);
      v16h bv = cat16(pb[0], pb[1]);
      c = __builtin_amdgcn_wmma_f32_16x16x32_f16(false, av, false, bv,
                                                 (short)0, c, false, false);
    }
    float bcv = bc[nt * 16 + n];
#pragma unroll
    for (int g = 0; g < 8; ++g) {
      float cv = c[g] + bcv;
      float lv = (float)l16[(mbase + g) * 256 + nt * 16 + n];
      acc8[g] += cv * lv;
    }
  }

  // reduce over the 16 lanes that share a row group (sum over N)
#pragma unroll
  for (int mask = 1; mask < 16; mask <<= 1) {
#pragma unroll
    for (int g = 0; g < 8; ++g)
      acc8[g] += __shfl_xor(acc8[g], mask, 32);
  }
  if (n == 0) { // lanes 0 (rows 0..7) and 16 (rows 8..15)
#pragma unroll
    for (int g = 0; g < 8; ++g)
      logits[(size_t)tile * 16 + mbase + g] = acc8[g];
  }
}

// ---- segment softmax: max, exp+sum, normalize*visited ----
__global__ void k_segmax(const float* __restrict__ logits,
                         const int* __restrict__ edges,
                         unsigned* __restrict__ mx) {
  int e = blockIdx.x * blockDim.x + threadIdx.x;
  if (e >= ETOT) return;
  int ni = edges[(size_t)e * 8 + 6];
  atomicMax(&mx[ni], f2ord(logits[e]));
}

__global__ void k_expden(const float* __restrict__ logits,
                         const int* __restrict__ edges,
                         const unsigned* __restrict__ mx,
                         float* __restrict__ exv, float* __restrict__ den) {
  int e = blockIdx.x * blockDim.x + threadIdx.x;
  if (e >= ETOT) return;
  int ni = edges[(size_t)e * 8 + 6];
  float m  = ord2f(mx[ni]);
  float ex = __expf(logits[e] - m);
  exv[e] = ex;
  atomicAdd(&den[ni], ex);
}

__global__ void k_target(const float* __restrict__ exv,
                         const int* __restrict__ edges,
                         const float* __restrict__ den,
                         const float* __restrict__ visited,
                         float* __restrict__ target) {
  int e = blockIdx.x * blockDim.x + threadIdx.x;
  if (e >= ETOT) return;
  int ni = edges[(size_t)e * 8 + 6];
  target[e] = exv[e] / (den[ni] + 1e-20f) * visited[ni];
}

// ---- per-batch top-K (radix select on order-mapped keys) + scatter-add ----
__global__ __launch_bounds__(256)
void k_topk(const float* __restrict__ target, const int* __restrict__ edges,
            float* __restrict__ out, const int* __restrict__ kptr) {
  __shared__ unsigned keys[EPER];
  __shared__ unsigned hist[16];
  __shared__ unsigned prefixSh, needSh, tieCnt;

  const int b = blockIdx.x, t = threadIdx.x;
  const int Kkeep = kptr[0];

  for (int i = t; i < EPER; i += 256)
    keys[i] = f2ord(target[(size_t)b * EPER + i]);
  if (t == 0) { prefixSh = 0u; needSh = (unsigned)Kkeep; tieCnt = 0u; }
  __syncthreads();

  for (int round = 0; round < 8; ++round) {
    int shift = 28 - round * 4;
    if (t < 16) hist[t] = 0u;
    __syncthreads();
    unsigned pref  = prefixSh;
    unsigned pmask = (round == 0) ? 0u : (0xFFFFFFFFu << (shift + 4));
    for (int i = t; i < EPER; i += 256) {
      unsigned u = keys[i];
      if ((u & pmask) == pref) atomicAdd(&hist[(u >> shift) & 15u], 1u);
    }
    __syncthreads();
    if (t == 0) {
      unsigned need = needSh, d = 0;
      for (int dd = 15; dd >= 0; --dd) {
        unsigned h = hist[dd];
        if (need > h) need -= h;
        else { d = (unsigned)dd; break; }
      }
      prefixSh = pref | (d << shift);
      needSh   = need;
    }
    __syncthreads();
  }

  const unsigned T = prefixSh;
  const unsigned need = needSh;
  for (int i = t; i < EPER; i += 256) {
    unsigned u = keys[i];
    bool take = false;
    if (u > T) take = true;
    else if (u == T) { unsigned p = atomicAdd(&tieCnt, 1u); take = (p < need); }
    if (take) {
      float v  = ord2f(u);
      int   nj = edges[((size_t)b * EPER + i) * 8 + 7];
      atomicAdd(&out[nj], v);
    }
  }
}

// ---------------------------------------------------------------------------
extern "C" void kernel_launch(void* const* d_in, const int* in_sizes, int n_in,
                              void* d_out, int out_size, void* d_ws, size_t ws_size,
                              hipStream_t stream) {
  const int*   edges   = (const int*)  d_in[0];
  const float* visited = (const float*)d_in[1];
  const float* nodeRep = (const float*)d_in[2];
  const float* relEmb  = (const float*)d_in[3];
  const float* qSrc    = (const float*)d_in[4];
  const float* qRel    = (const float*)d_in[5];
  const float* Wl      = (const float*)d_in[6];
  const float* bl      = (const float*)d_in[7];
  const float* Wr      = (const float*)d_in[8];
  const float* br      = (const float*)d_in[9];
  const float* Wc      = (const float*)d_in[10];
  const float* bc      = (const float*)d_in[11];
  const int*   kptr    = (const int*)  d_in[12];
  float* out = (float*)d_out;

  // workspace layout
  char* ws = (char*)d_ws;
  size_t off = 0;
  _Float16* WlP = (_Float16*)(ws + off); off += (size_t)512 * 256 * 2;
  _Float16* WrP = (_Float16*)(ws + off); off += (size_t)512 * 256 * 2;
  _Float16* WcP = (_Float16*)(ws + off); off += (size_t)256 * 256 * 2;
  float*    logits = (float*)(ws + off); off += (size_t)ETOT * 4;
  float*    exv    = (float*)(ws + off); off += (size_t)ETOT * 4;
  float*    target = (float*)(ws + off); off += (size_t)ETOT * 4;
  unsigned* mx     = (unsigned*)(ws + off); off += (size_t)NN * 4;
  float*    den    = (float*)(ws + off); off += (size_t)NN * 4;

  // 1. pack weights to f16 WMMA fragments
  k_pack<<<(512 * 256 + 255) / 256, 256, 0, stream>>>(Wl, WlP, 256, 16, 512 * 256);
  k_pack<<<(512 * 256 + 255) / 256, 256, 0, stream>>>(Wr, WrP, 256, 16, 512 * 256);
  k_pack<<<(256 * 256 + 255) / 256, 256, 0, stream>>>(Wc, WcP, 256, 8, 256 * 256);

  // 2. init output + softmax stats
  k_init<<<NN / 256, 256, 0, stream>>>(out, mx, den);

  // 3. WMMA logits (one wave per 16-edge tile)
  k_logits<<<ETOT / 16, 32, 0, stream>>>(edges, nodeRep, relEmb, qSrc, qRel,
                                         WlP, bl, WrP, br, WcP, bc, logits);

  // 4. segment softmax + target score
  k_segmax<<<ETOT / 256, 256, 0, stream>>>(logits, edges, mx);
  k_expden<<<ETOT / 256, 256, 0, stream>>>(logits, edges, mx, exv, den);
  k_target<<<ETOT / 256, 256, 0, stream>>>(exv, edges, den, visited, target);

  // 5. per-batch top-K + scatter into output
  k_topk<<<BQ, 256, 0, stream>>>(target, edges, out, kptr);
}
